// Model_24180665876773
// MI455X (gfx1250) — matrix operand order repaired and checked by reading, not compiled, not run on hardware
//
#include <hip/hip_runtime.h>
#include <math.h>

// ---------------------------------------------------------------------------
// FEDformer-style forward for MI455X (gfx1250, wave32).
// Big matmuls -> V_WMMA_F32_16X16X32_F16, register-blocked 16x64 per wave
// (4 accumulators, A-fragment reuse, 4 independent WMMAs per k-step).
// Small DFT / decomposition / norm kernels stay fp32 VALU (negligible cost).
// ---------------------------------------------------------------------------

typedef _Float16 half8  __attribute__((ext_vector_type(8)));
typedef _Float16 half16 __attribute__((ext_vector_type(16)));
typedef float    float8 __attribute__((ext_vector_type(8)));

#define TPB 256

// ---------------- generic f32 -> f16 convert with K padding ----------------
__global__ void k_cvt(const float* __restrict__ src, _Float16* __restrict__ dst,
                      int K, int Kp, long total) {
  long idx = (long)blockIdx.x * blockDim.x + threadIdx.x;
  if (idx >= total) return;
  int  j = (int)(idx % Kp);
  long r = idx / Kp;
  dst[idx] = (j < K) ? (_Float16)src[r * (long)K + j] : (_Float16)0.0f;
}

// ---------------- WMMA fragment loaders ------------------------------------
__device__ __forceinline__ half16 load_afrag(const _Float16* arow, int k) {
  half8 a0 = *(const half8*)(arow + k);
  half8 a1 = *(const half8*)(arow + k + 8);
  return __builtin_shufflevector(a0, a1, 0,1,2,3,4,5,6,7,8,9,10,11,12,13,14,15);
}
__device__ __forceinline__ half16 load_bfrag(const _Float16* brow, int k) {
  half8 b0 = *(const half8*)(brow + k);
  half8 b1 = *(const half8*)(brow + k + 8);
  return __builtin_shufflevector(b0, b1, 0,1,2,3,4,5,6,7,8,9,10,11,12,13,14,15);
}
__device__ __forceinline__ float gelu_f(float v) {
  return 0.5f * v * (1.f + erff(v * 0.70710678118654752f));
}

// ---------------- blocked WMMA GEMM: wave tile = 16 rows x 64 cols ---------
// C[M,N] = A[M,Kp] * W[N,Kp]^T (+bias).  Requires M%32==0, N%64==0.
// blockDim = (32,2): 2 waves -> 32 rows; grid = (N/64, M/32).
// A fragment: lane (m = lane&15) holds K=[g*16..g*16+15] (g = lane>>4)
// B fragment: lane holds 16 contiguous K for output feature n (row of W).
__global__ __launch_bounds__(64)
void k_gemm_wmma4(const _Float16* __restrict__ A, const _Float16* __restrict__ Bw,
                  const float* __restrict__ bias, float* __restrict__ C,
                  _Float16* __restrict__ C16, int M, int N, int Kp, int fuse_gelu) {
  const int lane = threadIdx.x;
  const int hi   = lane >> 4;
  const int lm   = lane & 15;
  const int tm   = (blockIdx.y * 2 + threadIdx.y) * 16;
  const int tn   = blockIdx.x * 64;

  const _Float16* arow = A  + (size_t)(tm + lm) * Kp + hi * 16;
  const _Float16* brow = Bw + (size_t)(tn + lm) * Kp + hi * 16;
  const size_t bstep = (size_t)16 * Kp;   // next 16 output features

  float8 acc0 = {0.f,0.f,0.f,0.f,0.f,0.f,0.f,0.f};
  float8 acc1 = acc0, acc2 = acc0, acc3 = acc0;

  for (int k = 0; k < Kp; k += 32) {
    half16 av = load_afrag(arow, k);
    half16 b0 = load_bfrag(brow,             k);
    half16 b1 = load_bfrag(brow + bstep,     k);
    half16 b2 = load_bfrag(brow + 2 * bstep, k);
    half16 b3 = load_bfrag(brow + 3 * bstep, k);
    acc0 = __builtin_amdgcn_wmma_f32_16x16x32_f16(false, av, false, b0, (short)0, acc0, false, false);
    acc1 = __builtin_amdgcn_wmma_f32_16x16x32_f16(false, av, false, b1, (short)0, acc1, false, false);
    acc2 = __builtin_amdgcn_wmma_f32_16x16x32_f16(false, av, false, b2, (short)0, acc2, false, false);
    acc3 = __builtin_amdgcn_wmma_f32_16x16x32_f16(false, av, false, b3, (short)0, acc3, false, false);
  }

  float8 accs[4] = {acc0, acc1, acc2, acc3};
#pragma unroll
  for (int j = 0; j < 4; ++j) {
    const int col = tn + j * 16 + lm;
    const float bb = bias ? bias[col] : 0.f;
#pragma unroll
    for (int g = 0; g < 8; ++g) {
      int row = tm + g + hi * 8;
      float v = accs[j][g] + bb;
      if (fuse_gelu) v = gelu_f(v);
      if (C)   C  [(size_t)row * N + col] = v;
      if (C16) C16[(size_t)row * N + col] = (_Float16)v;
    }
  }
}

// ---------------- fallback WMMA GEMM: one 16x16 tile per wave --------------
__global__ __launch_bounds__(32)
void k_gemm_wmma(const _Float16* __restrict__ A, const _Float16* __restrict__ Bw,
                 const float* __restrict__ bias, float* __restrict__ C,
                 _Float16* __restrict__ C16, int M, int N, int Kp, int fuse_gelu) {
  const int tn   = blockIdx.x * 16;
  const int tm   = blockIdx.y * 16;
  const int lane = threadIdx.x;
  const int hi   = lane >> 4;
  const int lm   = lane & 15;

  const _Float16* arow = A  + (size_t)(tm + lm) * Kp + hi * 16;
  const _Float16* brow = Bw + (size_t)(tn + lm) * Kp + hi * 16;

  float8 acc = {0.f,0.f,0.f,0.f,0.f,0.f,0.f,0.f};
  for (int k = 0; k < Kp; k += 32) {
    half16 av = load_afrag(arow, k);
    half16 bv = load_bfrag(brow, k);
    acc = __builtin_amdgcn_wmma_f32_16x16x32_f16(false, av, false, bv,
                                                 (short)0, acc, false, false);
  }
  const int col = tn + lm;
  const float bb = bias ? bias[col] : 0.f;
#pragma unroll
  for (int g = 0; g < 8; ++g) {
    int row = tm + g + hi * 8;
    float v = acc[g] + bb;
    if (fuse_gelu) v = gelu_f(v);
    if (C)   C  [(size_t)row * N + col] = v;
    if (C16) C16[(size_t)row * N + col] = (_Float16)v;
  }
}

// ------------- series_decomp on raw input [B,L,N] -> [B*N, L] --------------
__global__ void k_decomp_input(const float* __restrict__ x, float* __restrict__ seas,
                               float* __restrict__ trend, int Bv, int L, int Nv) {
  int idx = blockIdx.x * TPB + threadIdx.x;
  int total = Bv * Nv * L;
  if (idx >= total) return;
  int l = idx % L, bn = idx / L, b = bn / Nv, n = bn % Nv;
  const float* xb = x + (size_t)b * L * Nv + n;
  float s = 0.f;
  for (int j = -12; j <= 12; ++j) {
    int t = l + j; t = t < 0 ? 0 : (t >= L ? L - 1 : t);
    s += xb[(size_t)t * Nv];
  }
  float mm = s * (1.f / 25.f);
  float v  = xb[(size_t)l * Nv];
  seas[idx]  = v - mm;
  trend[idx] = mm;
}

// ------------- series_decomp along patch axis of [rows, np, D] -------------
__global__ void k_sd_patch(const float* __restrict__ src, float* __restrict__ dst,
                           int np, int D, long total) {
  long idx = (long)blockIdx.x * TPB + threadIdx.x;
  if (idx >= total) return;
  int d  = (int)(idx % D);
  int l  = (int)((idx / D) % np);
  long bn = idx / ((long)D * np);
  const float* base = src + bn * (long)np * D + d;
  float s = 0.f;
  for (int j = -12; j <= 12; ++j) {
    int t = l + j; t = t < 0 ? 0 : (t >= np ? np - 1 : t);
    s += base[(size_t)t * D];
  }
  dst[idx] = src[idx] - s * (1.f / 25.f);
}

// ------------- elementwise add: dst = a + b --------------------------------
__global__ void k_add2(float* __restrict__ dst, const float* __restrict__ a,
                       const float* __restrict__ b, long total) {
  long idx = (long)blockIdx.x * TPB + threadIdx.x;
  if (idx >= total) return;
  dst[idx] = a[idx] + b[idx];
}

// ------------- patch gather into f16 (K padded to 32) ----------------------
__global__ void k_patch_gather(const float* __restrict__ src, _Float16* __restrict__ dst,
                               int np, int srcOff, int validT, long total) {
  long idx = (long)blockIdx.x * TPB + threadIdx.x;
  if (idx >= total) return;
  int  j  = (int)(idx & 31);
  int  p  = (int)((idx >> 5) % np);
  long bn = idx / ((long)np * 32);
  float v = 0.f;
  if (j < 16) {
    int t = p * 8 + j;
    if (t < validT) v = src[bn * 512 + srcOff + t];
  }
  dst[idx] = (_Float16)v;
}

// ------------- X[r,d] += bias[d] + pos[r%np, d] ----------------------------
__global__ void k_add_bias_pos(float* __restrict__ X, const float* __restrict__ bias,
                               const float* __restrict__ pos, int np, long total) {
  long idx = (long)blockIdx.x * TPB + threadIdx.x;
  if (idx >= total) return;
  int d = (int)(idx % 512);
  long r = idx / 512;
  int p = (int)(r % np);
  X[idx] += bias[d] + pos[(size_t)p * 512 + d];
}

// ------------- LayerNorm over last dim (optional residual x2) --------------
__global__ __launch_bounds__(128)
void k_layernorm(const float* __restrict__ x, const float* __restrict__ x2,
                 const float* __restrict__ w, const float* __restrict__ b,
                 float* __restrict__ y, int D) {
  __shared__ float red[128];
  const long row = blockIdx.x;
  const float* xr  = x + row * (long)D;
  const float* x2r = x2 ? x2 + row * (long)D : nullptr;
  int tid = threadIdx.x;
  float s = 0.f;
  for (int i = tid; i < D; i += 128) { float v = xr[i] + (x2r ? x2r[i] : 0.f); s += v; }
  red[tid] = s; __syncthreads();
  for (int off = 64; off > 0; off >>= 1) { if (tid < off) red[tid] += red[tid + off]; __syncthreads(); }
  float mu = red[0] / D; __syncthreads();
  float s2 = 0.f;
  for (int i = tid; i < D; i += 128) { float v = xr[i] + (x2r ? x2r[i] : 0.f) - mu; s2 += v * v; }
  red[tid] = s2; __syncthreads();
  for (int off = 64; off > 0; off >>= 1) { if (tid < off) red[tid] += red[tid + off]; __syncthreads(); }
  float rstd = rsqrtf(red[0] / D + 1e-5f); __syncthreads();
  for (int i = tid; i < D; i += 128) {
    float v = xr[i] + (x2r ? x2r[i] : 0.f);
    y[row * (long)D + i] = (v - mu) * rstd * w[i] + b[i];
  }
}

// ------------- subtract per-(bn,d) mean over patch axis (in-place) ---------
__global__ void k_sub_patch_mean(float* __restrict__ x, int np, long total) {
  long idx = (long)blockIdx.x * TPB + threadIdx.x;
  if (idx >= total) return;
  int  d  = (int)(idx % 512);
  long bn = idx / 512;
  float* base = x + bn * (long)np * 512 + d;
  float s = 0.f;
  for (int l = 0; l < np; ++l) s += base[(size_t)l * 512];
  float mu = s / np;
  for (int l = 0; l < np; ++l) base[(size_t)l * 512] -= mu;
}

// ------------- forward DFT of q [BN*L, 512] -> xf [BN,8,64,Mm] -------------
__global__ void k_dft_fwd(const float* __restrict__ q, float* __restrict__ xr,
                          float* __restrict__ xi, int L, int Mm, long total) {
  long idx = (long)blockIdx.x * TPB + threadIdx.x;
  if (idx >= total) return;
  int  m  = (int)(idx % Mm);
  int  e  = (int)((idx / Mm) & 63);
  int  h  = (int)((idx / ((long)Mm * 64)) & 7);
  long bn = idx / ((long)Mm * 512);
  const float* base = q + bn * (long)L * 512 + h * 64 + e;
  float w = -6.2831853071795864769f * m / L;
  float sr = 0.f, si = 0.f;
  for (int l = 0; l < L; ++l) {
    float v = base[(size_t)l * 512];
    float ang = w * l;
    sr += v * cosf(ang); si += v * sinf(ang);
  }
  xr[idx] = sr; xi[idx] = si;
}

// ------------- per-mode complex mixing: of[o,m] = sum_e xf[e,m]*W[h,e,o,m] -
__global__ void k_mix(const float* __restrict__ xr, const float* __restrict__ xi,
                      const float* __restrict__ W, float* __restrict__ outr,
                      float* __restrict__ outi, int Mm, long total) {
  long idx = (long)blockIdx.x * TPB + threadIdx.x;
  if (idx >= total) return;
  int  m  = (int)(idx % Mm);
  int  o  = (int)((idx / Mm) & 63);
  int  h  = (int)((idx / ((long)Mm * 64)) & 7);
  long bn = idx / ((long)Mm * 512);
  const float* Wr = W;
  const float* Wi = W + (size_t)8 * 64 * 64 * Mm;
  float sr = 0.f, si = 0.f;
  for (int e = 0; e < 64; ++e) {
    long xidx = ((bn * 8 + h) * 64 + e) * (long)Mm + m;
    size_t widx = (((size_t)(h * 64 + e) * 64 + o) * Mm) + m;
    float ar = xr[xidx], ai = xi[xidx];
    float wr = Wr[widx], wi = Wi[widx];
    sr += ar * wr - ai * wi;
    si += ar * wi + ai * wr;
  }
  outr[idx] = sr; outi[idx] = si;
}

// ------------- inverse real DFT; output written with the .view quirk -------
__global__ void k_dft_inv(const float* __restrict__ ofr, const float* __restrict__ ofi,
                          float* __restrict__ a, int L, int Mm, float scale, long total) {
  long idx = (long)blockIdx.x * TPB + threadIdx.x;
  if (idx >= total) return;
  int  t  = (int)(idx % L);
  int  o  = (int)((idx / L) & 63);
  int  h  = (int)((idx / ((long)L * 64)) & 7);
  long bn = idx / ((long)L * 512);
  const float* rr = ofr + ((bn * 8 + h) * 64 + o) * (long)Mm;
  const float* ii = ofi + ((bn * 8 + h) * 64 + o) * (long)Mm;
  float s = rr[0];
  float w = 6.2831853071795864769f * t / L;
  for (int m = 1; m < Mm; ++m) {
    float ang = w * m;
    s += 2.f * (rr[m] * cosf(ang) - ii[m] * sinf(ang));
  }
  a[bn * (long)L * 512 + (long)(h * 64 + o) * L + t] = s * scale / L;
}

// ------------- cross attention: att[x,y] = ctanh(sum_e qf[e,x]*kf[e,y]) ----
__global__ void k_cross_att(const float* __restrict__ qr, const float* __restrict__ qi,
                            const float* __restrict__ kr, const float* __restrict__ ki,
                            float* __restrict__ ar, float* __restrict__ ai,
                            int Mq, int Mk, long total) {
  long idx = (long)blockIdx.x * TPB + threadIdx.x;
  if (idx >= total) return;
  int  y  = (int)(idx % Mk);
  int  x  = (int)((idx / Mk) % Mq);
  int  h  = (int)((idx / ((long)Mk * Mq)) & 7);
  long bn = idx / ((long)Mk * Mq * 8);
  float sr = 0.f, si = 0.f;
  for (int e = 0; e < 64; ++e) {
    long qidx = ((bn * 8 + h) * 64 + e) * (long)Mq + x;
    long kidx = ((bn * 8 + h) * 64 + e) * (long)Mk + y;
    float a0 = qr[qidx], b0 = qi[qidx], c0 = kr[kidx], d0 = ki[kidx];
    sr += a0 * c0 - b0 * d0;
    si += a0 * d0 + b0 * c0;
  }
  // complex tanh(sr + i si)
  float denom = coshf(2.f * sr) + cosf(2.f * si);
  ar[idx] = sinhf(2.f * sr) / denom;
  ai[idx] = sinf (2.f * si) / denom;
}

// ------------- v[e,x] = sum_y att[x,y] * kf[e,y] ---------------------------
__global__ void k_cross_v(const float* __restrict__ ar, const float* __restrict__ ai,
                          const float* __restrict__ kr, const float* __restrict__ ki,
                          float* __restrict__ vr, float* __restrict__ vi,
                          int Mq, int Mk, long total) {
  long idx = (long)blockIdx.x * TPB + threadIdx.x;
  if (idx >= total) return;
  int  x  = (int)(idx % Mq);
  int  e  = (int)((idx / Mq) & 63);
  int  h  = (int)((idx / ((long)Mq * 64)) & 7);
  long bn = idx / ((long)Mq * 512);
  float sr = 0.f, si = 0.f;
  for (int y = 0; y < Mk; ++y) {
    long aidx = ((bn * 8 + h) * (long)Mq + x) * Mk + y;
    long kidx = ((bn * 8 + h) * 64 + e) * (long)Mk + y;
    float a0 = ar[aidx], b0 = ai[aidx], c0 = kr[kidx], d0 = ki[kidx];
    sr += a0 * c0 - b0 * d0;
    si += a0 * d0 + b0 * c0;
  }
  vr[idx] = sr; vi[idx] = si;
}

// ------------- ofm[o,x] = sum_e v[e,x] * W[h,e,o,x] ------------------------
__global__ void k_cross_ofm(const float* __restrict__ vr, const float* __restrict__ vi,
                            const float* __restrict__ W, float* __restrict__ ofr,
                            float* __restrict__ ofi, int Mq, long total) {
  long idx = (long)blockIdx.x * TPB + threadIdx.x;
  if (idx >= total) return;
  int  x  = (int)(idx % Mq);
  int  o  = (int)((idx / Mq) & 63);
  int  h  = (int)((idx / ((long)Mq * 64)) & 7);
  long bn = idx / ((long)Mq * 512);
  const float* Wr = W;
  const float* Wi = W + (size_t)8 * 64 * 64 * Mq;
  float sr = 0.f, si = 0.f;
  for (int e = 0; e < 64; ++e) {
    long vidx = ((bn * 8 + h) * 64 + e) * (long)Mq + x;
    size_t widx = (((size_t)(h * 64 + e) * 64 + o) * Mq) + x;
    float a0 = vr[vidx], b0 = vi[vidx], c0 = Wr[widx], d0 = Wi[widx];
    sr += a0 * c0 - b0 * d0;
    si += a0 * d0 + b0 * c0;
  }
  ofr[idx] = sr; ofi[idx] = si;
}

// ------------- final: out[b,p,n] = head_tail + trend_proj ------------------
__global__ void k_final(const float* __restrict__ headout, const float* __restrict__ trendp,
                        float* __restrict__ out, int total) {
  int idx = blockIdx.x * TPB + threadIdx.x;
  if (idx >= total) return;
  int n = idx % 21;
  int p = (idx / 21) % 96;
  int b = idx / (21 * 96);
  long bn = (long)b * 21 + n;
  out[idx] = headout[bn * 272 + 176 + p] + trendp[bn * 96 + p];
}

// ===========================================================================
extern "C" void kernel_launch(void* const* d_in, const int* /*in_sizes*/, int /*n_in*/,
                              void* d_out, int /*out_size*/, void* d_ws, size_t /*ws_size*/,
                              hipStream_t stream) {
  // ---- inputs (setup_inputs order; marks are unused by the reference) ----
  const float* x_enc       = (const float*)d_in[0];
  const float* pe_enc_w    = (const float*)d_in[4];
  const float* pe_enc_b    = (const float*)d_in[5];
  const float* pe_enc_pos  = (const float*)d_in[6];
  const float* pe_dec_w    = (const float*)d_in[7];
  const float* pe_dec_b    = (const float*)d_in[8];
  const float* pe_dec_pos  = (const float*)d_in[9];
  const float* enc_q_w     = (const float*)d_in[10];
  const float* enc_q_b     = (const float*)d_in[11];
  const float* enc_o_w     = (const float*)d_in[12];
  const float* enc_o_b     = (const float*)d_in[13];
  const float* enc_four_w  = (const float*)d_in[14];
  const float* enc_c1_w    = (const float*)d_in[15];
  const float* enc_c1_b    = (const float*)d_in[16];
  const float* enc_c2_w    = (const float*)d_in[17];
  const float* enc_c2_b    = (const float*)d_in[18];
  const float* enc_norm_w  = (const float*)d_in[19];
  const float* enc_norm_b  = (const float*)d_in[20];
  const float* dec_sq_w    = (const float*)d_in[21];
  const float* dec_sq_b    = (const float*)d_in[22];
  const float* dec_so_w    = (const float*)d_in[23];
  const float* dec_so_b    = (const float*)d_in[24];
  const float* dec_sfour_w = (const float*)d_in[25];
  const float* dec_cq_w    = (const float*)d_in[26];
  const float* dec_cq_b    = (const float*)d_in[27];
  const float* dec_ck_w    = (const float*)d_in[28];
  const float* dec_ck_b    = (const float*)d_in[29];
  const float* dec_co_w    = (const float*)d_in[30];
  const float* dec_co_b    = (const float*)d_in[31];
  const float* dec_cfour_w = (const float*)d_in[32];
  const float* dec_c1_w    = (const float*)d_in[33];
  const float* dec_c1_b    = (const float*)d_in[34];
  const float* dec_c2_w    = (const float*)d_in[35];
  const float* dec_c2_b    = (const float*)d_in[36];
  const float* dec_ln_w    = (const float*)d_in[37];
  const float* dec_ln_b    = (const float*)d_in[38];
  const float* dec_norm_w  = (const float*)d_in[39];
  const float* dec_norm_b  = (const float*)d_in[40];
  const float* head_w      = (const float*)d_in[41];
  const float* head_b      = (const float*)d_in[42];
  const float* trend_w     = (const float*)d_in[43];
  const float* trend_b     = (const float*)d_in[44];
  float* out = (float*)d_out;

  const int L = 512, Dm = 512, DFF = 2048, NPE = 63, NPD = 17;
  const int ME = 31, MD = 8, BN = 672;
  const long MEnc = (long)BN * NPE;   // 42336 rows (div by 32)
  const long MDec = (long)BN * NPD;   // 11424 rows (div by 32)

  // ---- workspace bump allocator ----
  char* wp = (char*)d_ws;
  auto alloc = [&](size_t bytes) -> char* {
    char* r = wp; wp += (bytes + 255) & ~(size_t)255; return r;
  };

  // f16 weight copies
  _Float16* w_pe_enc = (_Float16*)alloc((size_t)512 * 32 * 2);
  _Float16* w_pe_dec = (_Float16*)alloc((size_t)512 * 32 * 2);
  _Float16* w_q[2], *w_o[2], *w_c1[2], *w_c2[2];
  for (int l = 0; l < 2; ++l) {
    w_q[l]  = (_Float16*)alloc((size_t)512 * 512 * 2);
    w_o[l]  = (_Float16*)alloc((size_t)512 * 512 * 2);
    w_c1[l] = (_Float16*)alloc((size_t)2048 * 512 * 2);
    w_c2[l] = (_Float16*)alloc((size_t)512 * 2048 * 2);
  }
  _Float16* w_dsq  = (_Float16*)alloc((size_t)512 * 512 * 2);
  _Float16* w_dso  = (_Float16*)alloc((size_t)512 * 512 * 2);
  _Float16* w_dcq  = (_Float16*)alloc((size_t)512 * 512 * 2);
  _Float16* w_dck  = (_Float16*)alloc((size_t)512 * 512 * 2);
  _Float16* w_dco  = (_Float16*)alloc((size_t)512 * 512 * 2);
  _Float16* w_dc1  = (_Float16*)alloc((size_t)2048 * 512 * 2);
  _Float16* w_dc2  = (_Float16*)alloc((size_t)512 * 2048 * 2);
  _Float16* w_head = (_Float16*)alloc((size_t)16 * 512 * 2);
  _Float16* w_trd  = (_Float16*)alloc((size_t)96 * 512 * 2);

  // activations
  float* seas   = (float*)alloc((size_t)BN * L * 4);
  float* trnd   = (float*)alloc((size_t)BN * L * 4);
  _Float16* t16 = (_Float16*)alloc((size_t)BN * L * 2);
  float* trendp = (float*)alloc((size_t)BN * 96 * 4);
  _Float16* patch16 = (_Float16*)alloc((size_t)MEnc * 32 * 2);
  float* X = (float*)alloc((size_t)MEnc * Dm * 4);
  float* T = (float*)alloc((size_t)MEnc * Dm * 4);
  float* S = (float*)alloc((size_t)MEnc * Dm * 4);
  _Float16* X16 = (_Float16*)alloc((size_t)MEnc * Dm * 2);
  _Float16* H16 = (_Float16*)alloc((size_t)MEnc * DFF * 2);
  // DFT scratch (sized for encoder; decoder fits inside)
  size_t fsz = (size_t)BN * 8 * 64 * ME * 4;
  float* xr  = (float*)alloc(fsz); float* xi  = (float*)alloc(fsz);
  float* ofr = (float*)alloc(fsz); float* ofi = (float*)alloc(fsz);
  // decoder
  float* Dx = (float*)alloc((size_t)MDec * Dm * 4);
  float* Dt = (float*)alloc((size_t)MDec * Dm * 4);
  float* Ds = (float*)alloc((size_t)MDec * Dm * 4);
  _Float16* D16  = (_Float16*)alloc((size_t)MDec * Dm * 2);
  _Float16* DH16 = (_Float16*)alloc((size_t)MDec * DFF * 2);
  size_t csz = (size_t)BN * 8 * 64 * MD * 4;
  float* qr = (float*)alloc(csz); float* qi = (float*)alloc(csz);
  float* kr = (float*)alloc(csz); float* ki = (float*)alloc(csz);
  float* vr = (float*)alloc(csz); float* vi = (float*)alloc(csz);
  float* atr = (float*)alloc((size_t)BN * 8 * MD * MD * 4);
  float* ati = (float*)alloc((size_t)BN * 8 * MD * MD * 4);
  float* headout = (float*)alloc((size_t)MDec * 16 * 4);

  // ---- helpers ----
  auto g1 = [](long n) { return dim3((unsigned)((n + TPB - 1) / TPB)); };
  auto cvt = [&](const float* s, _Float16* d, int K, int Kp, long rows) {
    long n = rows * (long)Kp;
    k_cvt<<<g1(n), TPB, 0, stream>>>(s, d, K, Kp, n);
  };
  auto gemm = [&](const _Float16* A, const _Float16* Bw, const float* bias,
                  float* C, _Float16* C16, long M, int N, int Kp, int gelu) {
    if ((N % 64 == 0) && (M % 32 == 0)) {
      k_gemm_wmma4<<<dim3(N / 64, (unsigned)(M / 32)), dim3(32, 2), 0, stream>>>(
          A, Bw, bias, C, C16, (int)M, N, Kp, gelu);
    } else {
      k_gemm_wmma<<<dim3(N / 16, (unsigned)(M / 16)), 32, 0, stream>>>(
          A, Bw, bias, C, C16, (int)M, N, Kp, gelu);
    }
  };

  // ---- weight conversion (f32 -> f16, patch-embed K padded 16->32) ----
  cvt(pe_enc_w, w_pe_enc, 16, 32, 512);
  cvt(pe_dec_w, w_pe_dec, 16, 32, 512);
  for (int l = 0; l < 2; ++l) {
    cvt(enc_q_w  + (size_t)l * 512 * 512,  w_q[l],  512, 512, 512);
    cvt(enc_o_w  + (size_t)l * 512 * 512,  w_o[l],  512, 512, 512);
    cvt(enc_c1_w + (size_t)l * 2048 * 512, w_c1[l], 512, 512, 2048);
    cvt(enc_c2_w + (size_t)l * 512 * 2048, w_c2[l], 2048, 2048, 512);
  }
  cvt(dec_sq_w, w_dsq, 512, 512, 512);
  cvt(dec_so_w, w_dso, 512, 512, 512);
  cvt(dec_cq_w, w_dcq, 512, 512, 512);
  cvt(dec_ck_w, w_dck, 512, 512, 512);
  cvt(dec_co_w, w_dco, 512, 512, 512);
  cvt(dec_c1_w, w_dc1, 512, 512, 2048);
  cvt(dec_c2_w, w_dc2, 2048, 2048, 512);
  cvt(head_w,  w_head, 512, 512, 16);
  cvt(trend_w, w_trd,  512, 512, 96);

  // ---- input decomposition + trend projection ----
  k_decomp_input<<<g1((long)BN * L), TPB, 0, stream>>>(x_enc, seas, trnd, 32, L, 21);
  cvt(trnd, t16, 512, 512, BN);
  gemm(t16, w_trd, trend_b, trendp, nullptr, BN, 96, 512, 0);

  // ---- encoder patch embedding ----
  k_patch_gather<<<g1(MEnc * 32), TPB, 0, stream>>>(seas, patch16, NPE, 0, 512, MEnc * 32);
  gemm(patch16, w_pe_enc, nullptr, X, nullptr, MEnc, 512, 32, 0);
  k_add_bias_pos<<<g1(MEnc * Dm), TPB, 0, stream>>>(X, pe_enc_b, pe_enc_pos, NPE, MEnc * Dm);

  // ---- encoder layers ----
  const long encAct = MEnc * Dm;
  for (int l = 0; l < 2; ++l) {
    cvt(X, X16, 512, 512, MEnc);
    gemm(X16, w_q[l], enc_q_b + (size_t)l * 512, T, nullptr, MEnc, 512, 512, 0);
    long tf = (long)BN * 8 * 64 * ME;
    k_dft_fwd<<<g1(tf), TPB, 0, stream>>>(T, xr, xi, NPE, ME, tf);
    k_mix<<<g1(tf), TPB, 0, stream>>>(xr, xi, enc_four_w + (size_t)l * 2 * 8 * 64 * 64 * ME,
                                      ofr, ofi, ME, tf);
    long ti = (long)BN * 8 * 64 * NPE;
    k_dft_inv<<<g1(ti), TPB, 0, stream>>>(ofr, ofi, T, NPE, ME, 1.f, ti);
    cvt(T, X16, 512, 512, MEnc);
    gemm(X16, w_o[l], enc_o_b + (size_t)l * 512, T, nullptr, MEnc, 512, 512, 0);
    k_add2<<<g1(encAct), TPB, 0, stream>>>(X, X, T, encAct);
    k_sd_patch<<<g1(encAct), TPB, 0, stream>>>(X, S, NPE, 512, encAct);
    // FFN with fused GELU, hidden kept f16-only
    cvt(S, X16, 512, 512, MEnc);
    gemm(X16, w_c1[l], enc_c1_b + (size_t)l * 2048, nullptr, H16, MEnc, 2048, 512, 1);
    gemm(H16, w_c2[l], enc_c2_b + (size_t)l * 512, T, nullptr, MEnc, 512, 2048, 0);
    k_add2<<<g1(encAct), TPB, 0, stream>>>(T, S, T, encAct);
    k_sd_patch<<<g1(encAct), TPB, 0, stream>>>(T, X, NPE, 512, encAct);
  }
  // my_Layernorm + subtract patch-mean -> enc_out in S
  k_layernorm<<<(unsigned)MEnc, 128, 0, stream>>>(X, (const float*)nullptr,
                                                  enc_norm_w, enc_norm_b, S, 512);
  k_sub_patch_mean<<<g1((long)BN * 512), TPB, 0, stream>>>(S, NPE, (long)BN * 512);

  // ---- decoder ----
  const long decAct = MDec * Dm;
  k_patch_gather<<<g1(MDec * 32), TPB, 0, stream>>>(seas, patch16, NPD, 512 - 48, 48, MDec * 32);
  gemm(patch16, w_pe_dec, nullptr, Dx, nullptr, MDec, 512, 32, 0);
  k_add_bias_pos<<<g1(decAct), TPB, 0, stream>>>(Dx, pe_dec_b, pe_dec_pos, NPD, decAct);

  // decoder self fourier attention
  cvt(Dx, D16, 512, 512, MDec);
  gemm(D16, w_dsq, dec_sq_b, Dt, nullptr, MDec, 512, 512, 0);
  long tfd = (long)BN * 8 * 64 * MD;
  k_dft_fwd<<<g1(tfd), TPB, 0, stream>>>(Dt, xr, xi, NPD, MD, tfd);
  k_mix<<<g1(tfd), TPB, 0, stream>>>(xr, xi, dec_sfour_w, ofr, ofi, MD, tfd);
  long tid_ = (long)BN * 8 * 64 * NPD;
  k_dft_inv<<<g1(tid_), TPB, 0, stream>>>(ofr, ofi, Dt, NPD, MD, 1.f, tid_);
  cvt(Dt, D16, 512, 512, MDec);
  gemm(D16, w_dso, dec_so_b, Ds, nullptr, MDec, 512, 512, 0);
  k_add2<<<g1(decAct), TPB, 0, stream>>>(Dx, Dx, Ds, decAct);
  k_layernorm<<<(unsigned)MDec, 128, 0, stream>>>(Dx, (const float*)nullptr,
                                                  dec_ln_w, dec_ln_b, Ds, 512);

  // cross fourier attention (q from Ds, k from enc_out S)
  cvt(Ds, D16, 512, 512, MDec);
  gemm(D16, w_dcq, dec_cq_b, Dt, nullptr, MDec, 512, 512, 0);
  k_dft_fwd<<<g1(tfd), TPB, 0, stream>>>(Dt, qr, qi, NPD, MD, tfd);
  cvt(S, X16, 512, 512, MEnc);
  gemm(X16, w_dck, dec_ck_b, T, nullptr, MEnc, 512, 512, 0);
  k_dft_fwd<<<g1(tfd), TPB, 0, stream>>>(T, kr, ki, NPE, MD, tfd);
  long ta = (long)BN * 8 * MD * MD;
  k_cross_att<<<g1(ta), TPB, 0, stream>>>(qr, qi, kr, ki, atr, ati, MD, MD, ta);
  k_cross_v<<<g1(tfd), TPB, 0, stream>>>(atr, ati, kr, ki, vr, vi, MD, MD, tfd);
  k_cross_ofm<<<g1(tfd), TPB, 0, stream>>>(vr, vi, dec_cfour_w, ofr, ofi, MD, tfd);
  k_dft_inv<<<g1(tid_), TPB, 0, stream>>>(ofr, ofi, Dt, NPD, MD,
                                          1.f / (512.f * 512.f), tid_);
  cvt(Dt, D16, 512, 512, MDec);
  gemm(D16, w_dco, dec_co_b, Dx, nullptr, MDec, 512, 512, 0);
  k_add2<<<g1(decAct), TPB, 0, stream>>>(Ds, Ds, Dx, decAct);
  k_layernorm<<<(unsigned)MDec, 128, 0, stream>>>(Ds, (const float*)nullptr,
                                                  dec_ln_w + 512, dec_ln_b + 512, Dx, 512);

  // decoder FFN + LN2 + final decoder LN
  cvt(Dx, D16, 512, 512, MDec);
  gemm(D16, w_dc1, dec_c1_b, nullptr, DH16, MDec, 2048, 512, 1);
  gemm(DH16, w_dc2, dec_c2_b, Dt, nullptr, MDec, 512, 2048, 0);
  k_layernorm<<<(unsigned)MDec, 128, 0, stream>>>(Dx, Dt, dec_ln_w + 1024,
                                                  dec_ln_b + 1024, Ds, 512);
  k_layernorm<<<(unsigned)MDec, 128, 0, stream>>>(Ds, (const float*)nullptr,
                                                  dec_norm_w, dec_norm_b, Dx, 512);

  // head + final assembly
  cvt(Dx, D16, 512, 512, MDec);
  gemm(D16, w_head, head_b, headout, nullptr, MDec, 16, 512, 0);
  int tot = 32 * 96 * 21;
  k_final<<<g1(tot), TPB, 0, stream>>>(headout, trendp, out, tot);
}
